// WindowAttention_73237782331844
// MI455X (gfx1250) — matrix operand order repaired and checked by reading, not compiled, not run on hardware
//
#include <hip/hip_runtime.h>

// ---------------- types ----------------
typedef __attribute__((ext_vector_type(8)))  _Float16 v8h;
typedef __attribute__((ext_vector_type(16))) _Float16 v16h;
typedef __attribute__((ext_vector_type(8)))  float    v8f;

#define DIM   256
#define TOK   64      // 8x8 window tokens
#define HEADS 8
#define HD    32

// LDS layout (bytes). Pool region [0, 104448) is reused as f32 output stage at proj time.
#define XW_LD 264   // f16 elems/row, 64 rows   -> 33792 B  (x window, token-major)
#define QK_LD 264   // f16 elems/row, 64 rows   -> 33792 B  (q[0..128) | k[128..256) for head quad)
#define VT_LD 72    // f16 elems/row, 128 rows  -> 18432 B  (v transposed: [hd][token], quad)
#define P_LD  72    // f16 elems/row, 16 rows x 8 waves -> 18432 B
#define OB_LD 264   // f16 elems/row, 64 rows   -> 33792 B  (attention output, persists)
#define OS_LD 260   // f32 elems/row, 64 rows   -> 66560 B  (overlays pool)

#define OFF_QK  33792
#define OFF_VT  67584
#define OFF_P   86016
#define OFF_OB  104448
#define SMEM_TOTAL (104448 + 33792)

// ---------------- WMMA fragment helpers (CDNA5 16x16x32 f16 layouts) ----------------
__device__ __forceinline__ v16h pack16(v8h lo, v8h hi) {
  v16h r;
#pragma unroll
  for (int i = 0; i < 8; ++i) { r[i] = lo[i]; r[8 + i] = hi[i]; }
  return r;
}

// A-matrix 16x32: lane m(0..15)/m+16 holds row m; K = kh*8+[0..7] then 16+kh*8+[0..7]
__device__ __forceinline__ v16h ldA(const _Float16* base, int ld, int r0, int c0, int lane) {
  const int m = lane & 15, kh = lane >> 4;
  const _Float16* p = base + (r0 + m) * ld + c0 + kh * 8;
  v8h lo = *(const v8h*)p;
  v8h hi = *(const v8h*)(p + 16);
  return pack16(lo, hi);
}

// B-matrix 32x16 from row-major [N][K]: lane n holds column n; K = kh*8+[0..7] then 16+kh*8+[0..7]
__device__ __forceinline__ v16h ldB(const _Float16* base, int ld, int n0, int k0, int lane) {
  const int n = lane & 15, kb = (lane >> 4) * 8;
  const _Float16* p = base + (n0 + n) * ld + k0 + kb;
  v8h lo = *(const v8h*)p;
  v8h hi = *(const v8h*)(p + 16);
  return pack16(lo, hi);
}

__device__ __forceinline__ v8f wmma_f16(v16h a, v16h b, v8f c) {
  return __builtin_amdgcn_wmma_f32_16x16x32_f16(false, a, false, b, (short)0, c, false, false);
}

__device__ __forceinline__ v8f v8f_zero() {
  v8f z;
#pragma unroll
  for (int i = 0; i < 8; ++i) z[i] = 0.0f;
  return z;
}

// ---------------- weight f32 -> f16 pre-conversion ----------------
__global__ void cvt_weights(const float* __restrict__ qkv_w, const float* __restrict__ proj_w,
                            _Float16* __restrict__ w16) {
  const int i = blockIdx.x * blockDim.x + threadIdx.x;   // 0 .. 196607
  w16[i] = (_Float16)qkv_w[i];
  if (i < DIM * DIM) w16[3 * DIM * DIM + i] = (_Float16)proj_w[i];
}

// ---------------- fused window attention ----------------
__global__ __launch_bounds__(256) void wattn_kernel(
    const float* __restrict__ x,
    const _Float16* __restrict__ qkvw16,   // [768][256] f16
    const float* __restrict__ qkvb,
    const _Float16* __restrict__ projw16,  // [256][256] f16
    const float* __restrict__ projb,
    float* __restrict__ out) {
  __shared__ __align__(16) unsigned char smem[SMEM_TOTAL];
  _Float16* xw = (_Float16*)(smem);
  _Float16* qk = (_Float16*)(smem + OFF_QK);
  _Float16* vT = (_Float16*)(smem + OFF_VT);
  _Float16* Pb = (_Float16*)(smem + OFF_P);
  _Float16* Ob = (_Float16*)(smem + OFF_OB);
  float* ostage = (float*)(smem);          // overlays xw/qk after last quad

  const int tid  = threadIdx.x;
  const int lane = tid & 31;
  const int wave = tid >> 5;
  const int half = lane >> 4;
  const int ln   = lane & 15;

  const int wid = blockIdx.x;              // 4096 windows
  const int b  = wid >> 10;
  const int wh = (wid >> 5) & 31;
  const int ww = wid & 31;
  const int h0 = wh * 8, w0 = ww * 8;

  // ---- stage x window as f16, token-major [64][256] ----
  {
    const int c = tid;  // one channel per thread
    const float* xp = x + (((size_t)(b * DIM + c) * 256 + h0) * 256 + w0);
#pragma unroll
    for (int i = 0; i < 8; ++i) {
      float4 a0 = *(const float4*)(xp + (size_t)i * 256);
      float4 a1 = *(const float4*)(xp + (size_t)i * 256 + 4);
      _Float16* d = xw + (i * 8) * XW_LD + c;
      d[0 * XW_LD] = (_Float16)a0.x; d[1 * XW_LD] = (_Float16)a0.y;
      d[2 * XW_LD] = (_Float16)a0.z; d[3 * XW_LD] = (_Float16)a0.w;
      d[4 * XW_LD] = (_Float16)a1.x; d[5 * XW_LD] = (_Float16)a1.y;
      d[6 * XW_LD] = (_Float16)a1.z; d[7 * XW_LD] = (_Float16)a1.w;
    }
  }

  const float scale = 0.17677669529663687f;  // 1/sqrt(32)

  for (int q4 = 0; q4 < 2; ++q4) {           // head quads (4 heads each)
    __syncthreads();                         // xw ready / previous quad's reads done

    // ---- QKV slice GEMM for quad: 64 x 384 = (q 128 | k 128 | v 128), K=256 ----
    // Each wave owns exactly one q tile, one k tile, one v tile (perfect balance),
    // sharing every A-fragment load across the 3 WMMAs.
    {
      const int rq = 128 * q4 + 16 * wave;          // q weight rows
      const int rk = 256 + 128 * q4 + 16 * wave;    // k weight rows
      const int rv = 512 + 128 * q4 + 16 * wave;    // v weight rows
      v8f accq[4], acck[4], accv[4];
#pragma unroll
      for (int mt = 0; mt < 4; ++mt) { accq[mt] = v8f_zero(); acck[mt] = v8f_zero(); accv[mt] = v8f_zero(); }
      for (int k0 = 0; k0 < DIM; k0 += 32) {
        v16h Bq = ldB(qkvw16, DIM, rq, k0, lane);
        v16h Bk = ldB(qkvw16, DIM, rk, k0, lane);
        v16h Bv = ldB(qkvw16, DIM, rv, k0, lane);
#pragma unroll
        for (int mt = 0; mt < 4; ++mt) {
          v16h Af = ldA(xw, XW_LD, mt * 16, k0, lane);
          accq[mt] = wmma_f16(Af, Bq, accq[mt]);
          acck[mt] = wmma_f16(Af, Bk, acck[mt]);
          accv[mt] = wmma_f16(Af, Bv, accv[mt]);
        }
      }
      const float bq = qkvb[rq + ln];
      const float bk = qkvb[rk + ln];
      const float bv = qkvb[rv + ln];
#pragma unroll
      for (int mt = 0; mt < 4; ++mt)
#pragma unroll
        for (int r = 0; r < 8; ++r) {
          const int row = mt * 16 + half * 8 + r;
          qk[row * QK_LD + 16 * wave + ln]       = (_Float16)(accq[mt][r] + bq);
          qk[row * QK_LD + 128 + 16 * wave + ln] = (_Float16)(acck[mt][r] + bk);
          vT[(16 * wave + ln) * VT_LD + row]     = (_Float16)(accv[mt][r] + bv);  // transposed
        }
    }
    __syncthreads();

    // ---- attention: 16 tasks (head-in-quad h4, row-block mt), 2 per wave ----
#pragma unroll
    for (int t = 0; t < 2; ++t) {
      const int task = wave * 2 + t;
      const int h4 = task >> 2;
      const int mt = task & 3;
      v16h Qf = ldA(qk, QK_LD, mt * 16, h4 * 32, lane);
      v8f S[4];
#pragma unroll
      for (int nt = 0; nt < 4; ++nt) S[nt] = v8f_zero();
#pragma unroll
      for (int nt = 0; nt < 4; ++nt) {
        v16h Kf = ldB(qk + 128 + h4 * 32, QK_LD, nt * 16, 0, lane);  // B[k][n] = K[token n][k]
        S[nt] = wmma_f16(Qf, Kf, S[nt]);
      }
      // softmax per row (row = VGPR index + 8*half; cols across 16 lanes + 4 tiles)
#pragma unroll
      for (int r = 0; r < 8; ++r) {
        float mx = -3.0e38f;
#pragma unroll
        for (int nt = 0; nt < 4; ++nt) { S[nt][r] *= scale; mx = fmaxf(mx, S[nt][r]); }
#pragma unroll
        for (int m = 1; m < 16; m <<= 1) mx = fmaxf(mx, __shfl_xor(mx, m, 32));
        float sum = 0.0f;
#pragma unroll
        for (int nt = 0; nt < 4; ++nt) { float e = __expf(S[nt][r] - mx); S[nt][r] = e; sum += e; }
#pragma unroll
        for (int m = 1; m < 16; m <<= 1) sum += __shfl_xor(sum, m, 32);
        const float inv = 1.0f / sum;
#pragma unroll
        for (int nt = 0; nt < 4; ++nt) S[nt][r] *= inv;
      }
      // stage P (wave-private) to convert C/D layout -> A layout
      _Float16* Pw = Pb + wave * 16 * P_LD;
#pragma unroll
      for (int nt = 0; nt < 4; ++nt)
#pragma unroll
        for (int r = 0; r < 8; ++r)
          Pw[(half * 8 + r) * P_LD + nt * 16 + ln] = (_Float16)S[nt][r];

      v16h PA0 = ldA(Pw, P_LD, 0, 0, lane);
      v16h PA1 = ldA(Pw, P_LD, 0, 32, lane);
#pragma unroll
      for (int nto = 0; nto < 2; ++nto) {    // O = P @ V (V via vT: contiguous over tokens)
        v8f Oa = v8f_zero();
        Oa = wmma_f16(PA0, ldB(vT, VT_LD, h4 * 32 + nto * 16, 0, lane), Oa);
        Oa = wmma_f16(PA1, ldB(vT, VT_LD, h4 * 32 + nto * 16, 32, lane), Oa);
        const int ocb = (4 * q4 + h4) * HD + nto * 16;
#pragma unroll
        for (int r = 0; r < 8; ++r)
          Ob[(mt * 16 + half * 8 + r) * OB_LD + ocb + ln] = (_Float16)Oa[r];
      }
    }
  }
  __syncthreads();

  // ---- proj GEMM: out = O(64x256) @ proj_w^T + b; 2 N-tiles per wave, shared A loads ----
  {
    const int n0 = 16 * wave;
    const int n1 = 16 * wave + 128;
    v8f acc0[4], acc1[4];
#pragma unroll
    for (int mt = 0; mt < 4; ++mt) { acc0[mt] = v8f_zero(); acc1[mt] = v8f_zero(); }
    for (int k0 = 0; k0 < DIM; k0 += 32) {
      v16h B0 = ldB(projw16, DIM, n0, k0, lane);
      v16h B1 = ldB(projw16, DIM, n1, k0, lane);
#pragma unroll
      for (int mt = 0; mt < 4; ++mt) {
        v16h Af = ldA(Ob, OB_LD, mt * 16, k0, lane);
        acc0[mt] = wmma_f16(Af, B0, acc0[mt]);
        acc1[mt] = wmma_f16(Af, B1, acc1[mt]);
      }
    }
    const float bv0 = projb[n0 + ln];
    const float bv1 = projb[n1 + ln];
#pragma unroll
    for (int mt = 0; mt < 4; ++mt)
#pragma unroll
      for (int r = 0; r < 8; ++r) {
        const int row = mt * 16 + half * 8 + r;
        ostage[row * OS_LD + n0 + ln] = acc0[mt][r] + bv0;
        ostage[row * OS_LD + n1 + ln] = acc1[mt][r] + bv1;
      }
  }
  __syncthreads();

  // ---- coalesced writeback to [B,C,H,W] ----
  {
    const int c = tid;
    float* op = out + (((size_t)(b * DIM + c) * 256 + h0) * 256 + w0);
#pragma unroll
    for (int i = 0; i < 8; ++i) {
      const float* s = ostage + (i * 8) * OS_LD + c;
      float4 a0, a1;
      a0.x = s[0 * OS_LD]; a0.y = s[1 * OS_LD]; a0.z = s[2 * OS_LD]; a0.w = s[3 * OS_LD];
      a1.x = s[4 * OS_LD]; a1.y = s[5 * OS_LD]; a1.z = s[6 * OS_LD]; a1.w = s[7 * OS_LD];
      *(float4*)(op + (size_t)i * 256) = a0;
      *(float4*)(op + (size_t)i * 256 + 4) = a1;
    }
  }
}

extern "C" void kernel_launch(void* const* d_in, const int* in_sizes, int n_in,
                              void* d_out, int out_size, void* d_ws, size_t ws_size,
                              hipStream_t stream) {
  const float* x      = (const float*)d_in[0];
  const float* qkv_w  = (const float*)d_in[1];
  const float* qkv_b  = (const float*)d_in[2];
  const float* proj_w = (const float*)d_in[3];
  const float* proj_b = (const float*)d_in[4];
  float* out = (float*)d_out;

  _Float16* w16 = (_Float16*)d_ws;               // 196608 (qkv) + 65536 (proj) halves = 512 KB
  cvt_weights<<<768, 256, 0, stream>>>(qkv_w, proj_w, w16);

  wattn_kernel<<<4096, 256, 0, stream>>>(x, w16, qkv_b, w16 + 3 * DIM * DIM, proj_b, out);
}
